// Att6_31490700214514
// MI455X (gfx1250) — compile-verified, matrix operand order checked by reading, not run on hardware
//
#include <hip/hip_runtime.h>
#include <hip/hip_bf16.h>
#include <math.h>

// ---------------------------------------------------------------------------
// Problem dims (fixed by the reference)
// ---------------------------------------------------------------------------
#define B_DIM 32
#define T_DIM 2048
#define D_DIM 1024
#define H_DIM 1024

// Main GEMM tiling
#define M_TILE 128              // rows of flattened (B*T) per workgroup
#define N_TILE 16               // H columns per inner tile
#define K_STEP 32               // bf16 WMMA K
#define A_ROW_BF16 D_DIM        // bf16 elements per A row in LDS (2048 B)
#define B_COL_BF16 (D_DIM + 16) // padded column stride (bank-conflict pad), 32B aligned

typedef __attribute__((ext_vector_type(16))) __bf16 v16bf;
typedef __attribute__((ext_vector_type(8)))  float  v8f;

// ---------------------------------------------------------------------------
// Branch-free tanh.
// gfx1250 has a hardware V_TANH_F32 transcendental; prefer the builtin.
// Fallback: tanh(x) = 1 - 2/(exp(2x)+1) -- exact limits at +-inf, no EXEC
// divergence (the libm tanhf lowers to a branchy saveexec region).
// ---------------------------------------------------------------------------
__device__ __forceinline__ float fast_tanhf(float x)
{
#if __has_builtin(__builtin_amdgcn_tanhf)
    return __builtin_amdgcn_tanhf(x);
#else
    const float e = __expf(2.0f * x);
    return 1.0f - 2.0f / (e + 1.0f);
#endif
}

// ---------------------------------------------------------------------------
// Kernel 1: g[b,h] = tanh(a@wa)[b,h] * tanh(b@wb)[b,h] * wh[h]
// Tiny (0.13 GFLOP): plain VALU, coalesced over h, scalar-broadcast a/b.
// ---------------------------------------------------------------------------
__global__ __launch_bounds__(256)
void Att6_gates_kernel(const float* __restrict__ a, const float* __restrict__ b,
                       const float* __restrict__ wa, const float* __restrict__ wb,
                       const float* __restrict__ wh, float* __restrict__ g)
{
    const int batch = blockIdx.x;
    const int tid = threadIdx.x;
    float acca[4] = {0.f, 0.f, 0.f, 0.f};
    float accb[4] = {0.f, 0.f, 0.f, 0.f};
    for (int d = 0; d < D_DIM; ++d) {
        const float av = a[batch * D_DIM + d];   // wave-uniform -> s_load
        const float bv = b[batch * D_DIM + d];
        const float* war = wa + (size_t)d * H_DIM;
        const float* wbr = wb + (size_t)d * H_DIM;
#pragma unroll
        for (int j = 0; j < 4; ++j) {
            acca[j] = fmaf(av, war[tid + 256 * j], acca[j]);
            accb[j] = fmaf(bv, wbr[tid + 256 * j], accb[j]);
        }
    }
#pragma unroll
    for (int j = 0; j < 4; ++j) {
        const int h = tid + 256 * j;
        g[batch * H_DIM + h] = fast_tanhf(acca[j]) * fast_tanhf(accb[j]) * wh[h];
    }
}

// ---------------------------------------------------------------------------
// Kernel 2: fused   att_raw[b,t] = sum_h tanh( (t @ wt)[b,t,h] ) * g[b,h]
// 137 GFLOP via v_wmma_f32_16x16x32_bf16. One 128-row tile per workgroup.
//   LDS: A strip 128x1024 bf16 = 256 KB  +  B tile 16x(1024+16) bf16 = 32.5 KB
//   (fits CDNA5's 320 KB WGP LDS; one WG per WGP, 8 waves)
// ---------------------------------------------------------------------------
__global__ __launch_bounds__(256)
void Att6_score_gemm_kernel(const float* __restrict__ t,
                            const float* __restrict__ wt,
                            const float* __restrict__ g,
                            float* __restrict__ att_raw)
{
    __shared__ __bf16 Alds[M_TILE * A_ROW_BF16];   // 262144 B
    __shared__ __bf16 Blds[N_TILE * B_COL_BF16];   //  33280 B (transposed: [n][k])

    const int tid  = threadIdx.x;
    const int wave = tid >> 5;         // 0..7, owns rows wave*16..wave*16+15
    const int lane = tid & 31;
    const int row0 = blockIdx.x * M_TILE;          // flattened b*T + t base
    const int batch = row0 / T_DIM;                // tile never crosses batch (2048%128==0)

    // ---- Stage A: 128 rows x 1024 f32 -> bf16 in LDS (f32->bf16 on the fly) ----
    {
        const float4* src = (const float4*)(t + (size_t)row0 * D_DIM);
        for (int r = 0; r < M_TILE; ++r) {
            const float4 v = src[(size_t)r * (D_DIM / 4) + tid];
            union { __bf16 h[4]; uint2 u; } p;
            p.h[0] = (__bf16)v.x; p.h[1] = (__bf16)v.y;
            p.h[2] = (__bf16)v.z; p.h[3] = (__bf16)v.w;
            *(uint2*)&Alds[r * A_ROW_BF16 + tid * 4] = p.u;   // ds_store_b64
        }
    }
    __syncthreads();

    // A fragment addressing in contiguous k order (B uses the same):
    //   lanes 0-15 -> rows, K=khalf..khalf+15 packed 2/VGPR; lanes 16-31 -> K+16
    const int mrow  = wave * 16 + (lane & 15);
    const int khalf = (lane >> 4) * 16;
    const v16bf* Abase = (const v16bf*)&Alds[mrow * A_ROW_BF16 + khalf];
    const v16bf* Bbase = (const v16bf*)&Blds[(lane & 15) * B_COL_BF16 + khalf];

    float attacc[8] = {0.f, 0.f, 0.f, 0.f, 0.f, 0.f, 0.f, 0.f};

    for (int n0 = 0; n0 < H_DIM; n0 += N_TILE) {
        // ---- Stage B tile: wt[0:1024, n0:n0+16] -> Blds[n][k] (bf16, transposed) ----
        // Each thread handles 32 (k-pair, n) cells; global reads coalesced over n,
        // LDS writes packed as b32 (two consecutive k per column).
#pragma unroll 4
        for (int j = 0; j < 32; ++j) {
            const int idx = tid + 256 * j;       // 0..8191
            const int n = idx & 15;
            const int k = (idx >> 4) * 2;
            const float v0 = wt[(size_t)k * H_DIM + n0 + n];
            const float v1 = wt[(size_t)(k + 1) * H_DIM + n0 + n];
            union { __bf16 h[2]; unsigned int u; } p;
            p.h[0] = (__bf16)v0; p.h[1] = (__bf16)v1;
            *(unsigned int*)&Blds[n * B_COL_BF16 + k] = p.u;
        }
        __syncthreads();   // B tile visible to all waves

        // ---- 16x16 output tile: K=1024 in 32 WMMA steps ----
        v8f c = {};
#pragma unroll 8
        for (int k0 = 0; k0 < D_DIM / K_STEP; ++k0) {
            const v16bf af = Abase[k0 * 2];      // 32 contiguous bytes of LDS
            const v16bf bf = Bbase[k0 * 2];
            c = __builtin_amdgcn_wmma_f32_16x16x32_bf16(
                    false, af, false, bf, (short)0, c, false, false);
        }

        // ---- Epilogue: tanh, scale by g, fold the 16 columns into att partials.
        // C layout: lane l holds column (l&15); VGPR r holds row r + 8*(l>>4).
        const float gv = g[batch * H_DIM + n0 + (lane & 15)];
#pragma unroll
        for (int r = 0; r < 8; ++r)
            attacc[r] = fmaf(fast_tanhf(c[r]), gv, attacc[r]);

        __syncthreads();   // done reading Blds before next overwrite
    }

    // ---- Reduce the 16 columns held across lanes of each half-wave ----
#pragma unroll
    for (int r = 0; r < 8; ++r) {
        float v = attacc[r];
        v += __shfl_xor(v, 1, 32);
        v += __shfl_xor(v, 2, 32);
        v += __shfl_xor(v, 4, 32);
        v += __shfl_xor(v, 8, 32);
        attacc[r] = v;
    }
    if ((lane & 15) == 0) {
        const int mbase = wave * 16 + (lane >> 4) * 8;  // lane0 -> rows 0-7, lane16 -> rows 8-15
#pragma unroll
        for (int r = 0; r < 8; ++r)
            att_raw[row0 + mbase + r] = attacc[r];
    }
}

// ---------------------------------------------------------------------------
// Kernel 3: softmax + mask + renormalize, algebraically fused:
//   att_i = exp(s_i - max) * m_i / sum_j exp(s_j - max) * m_j   (Z cancels)
// In-place over att rows; one block per batch. Mask is JAX bool (1 byte/elem).
// ---------------------------------------------------------------------------
__global__ __launch_bounds__(256)
void Att6_softmax_mask_kernel(float* __restrict__ att, const unsigned char* __restrict__ mask)
{
    __shared__ float red[8];
    const int b = blockIdx.x;
    const int tid = threadIdx.x;
    float* row = att + (size_t)b * T_DIM;
    const unsigned char* mr = mask + (size_t)b * T_DIM;

    float v[8], mk[8];
    float mx = -3.402823466e38f;
#pragma unroll
    for (int j = 0; j < 8; ++j) {
        const int i = tid + 256 * j;
        v[j] = row[i];
        mk[j] = mr[i] ? 1.0f : 0.0f;
        mx = fmaxf(mx, v[j]);
    }
#pragma unroll
    for (int off = 16; off; off >>= 1) mx = fmaxf(mx, __shfl_xor(mx, off, 32));
    if ((tid & 31) == 0) red[tid >> 5] = mx;
    __syncthreads();
    float bmax = red[0];
#pragma unroll
    for (int w = 1; w < 8; ++w) bmax = fmaxf(bmax, red[w]);
    __syncthreads();

    float s = 0.f;
#pragma unroll
    for (int j = 0; j < 8; ++j) { v[j] = __expf(v[j] - bmax) * mk[j]; s += v[j]; }
#pragma unroll
    for (int off = 16; off; off >>= 1) s += __shfl_xor(s, off, 32);
    if ((tid & 31) == 0) red[tid >> 5] = s;
    __syncthreads();
    float total = 0.f;
#pragma unroll
    for (int w = 0; w < 8; ++w) total += red[w];
    const float inv = 1.0f / total;
#pragma unroll
    for (int j = 0; j < 8; ++j) row[tid + 256 * j] = v[j] * inv;
}

// ---------------------------------------------------------------------------
// Kernel 4/5: out[b,d] = sum_t t[b,t,d] * att[b,t]
// Deterministic (no float atomics): 16 T-chunks write partials to workspace,
// then a fixed-order reduce writes d_out fully.
// ---------------------------------------------------------------------------
#define T_CHUNKS 16
__global__ __launch_bounds__(256)
void Att6_pool_partial_kernel(const float* __restrict__ t, const float* __restrict__ att,
                              float* __restrict__ partial)
{
    const int batch = blockIdx.x;
    const int chunk = blockIdx.y;
    const int tid = threadIdx.x;              // covers d = tid*4 .. tid*4+3
    const int TCH = T_DIM / T_CHUNKS;         // 128
    const float4* tp = (const float4*)(t + ((size_t)batch * T_DIM + (size_t)chunk * TCH) * D_DIM);
    const float* ap = att + (size_t)batch * T_DIM + chunk * TCH;

    float4 acc = {0.f, 0.f, 0.f, 0.f};
    for (int tt = 0; tt < TCH; ++tt) {
        __builtin_prefetch(&tp[(size_t)(tt + 2) * (D_DIM / 4) + tid], 0, 0);
        const float w = ap[tt];
        const float4 x = tp[(size_t)tt * (D_DIM / 4) + tid];
        acc.x = fmaf(x.x, w, acc.x);
        acc.y = fmaf(x.y, w, acc.y);
        acc.z = fmaf(x.z, w, acc.z);
        acc.w = fmaf(x.w, w, acc.w);
    }
    float4* out4 = (float4*)(partial + ((size_t)chunk * B_DIM + batch) * D_DIM);
    out4[tid] = acc;
}

__global__ __launch_bounds__(256)
void Att6_pool_reduce_kernel(const float* __restrict__ partial, float* __restrict__ out)
{
    const int idx = blockIdx.x * 256 + threadIdx.x;   // 0 .. B*D-1
    float s = 0.f;
#pragma unroll
    for (int c = 0; c < T_CHUNKS; ++c)
        s += partial[(size_t)c * B_DIM * D_DIM + idx];
    out[idx] = s;
}

// ---------------------------------------------------------------------------
// Launcher.  Inputs (dict order): t, a, b, mask, wt, wa, wb, wh.
// Workspace layout: g (128 KB) | att (256 KB) | partial (2 MB)  => ~2.5 MB.
// ---------------------------------------------------------------------------
extern "C" void kernel_launch(void* const* d_in, const int* in_sizes, int n_in,
                              void* d_out, int out_size, void* d_ws, size_t ws_size,
                              hipStream_t stream)
{
    (void)in_sizes; (void)n_in; (void)out_size; (void)ws_size;
    const float* t  = (const float*)d_in[0];
    const float* a  = (const float*)d_in[1];
    const float* b  = (const float*)d_in[2];
    const unsigned char* mask = (const unsigned char*)d_in[3];  // JAX bool: 1 byte/elem
    const float* wt = (const float*)d_in[4];
    const float* wa = (const float*)d_in[5];
    const float* wb = (const float*)d_in[6];
    const float* wh = (const float*)d_in[7];
    float* out = (float*)d_out;

    char* ws = (char*)d_ws;
    float* g       = (float*)(ws);                       // B*H      = 131072 B
    float* att     = (float*)(ws + (size_t)131072);      // B*T      = 262144 B
    float* partial = (float*)(ws + (size_t)393216);      // 16*B*D   = 2097152 B

    Att6_gates_kernel<<<B_DIM, 256, 0, stream>>>(a, b, wa, wb, wh, g);
    Att6_score_gemm_kernel<<<(B_DIM * T_DIM) / M_TILE, 256, 0, stream>>>(t, wt, g, att);
    Att6_softmax_mask_kernel<<<B_DIM, 256, 0, stream>>>(att, mask);
    Att6_pool_partial_kernel<<<dim3(B_DIM, T_CHUNKS), 256, 0, stream>>>(t, att, partial);
    Att6_pool_reduce_kernel<<<(B_DIM * D_DIM) / 256, 256, 0, stream>>>(partial, out);
}
